// LearnableVisitEncoder_76888504533105
// MI455X (gfx1250) — matrix operand order repaired and checked by reading, not compiled, not run on hardware
//
#include <hip/hip_runtime.h>

typedef unsigned int u32;
typedef __attribute__((ext_vector_type(16))) __bf16 v16bf;
typedef __attribute__((ext_vector_type(8)))  float  v8f;

#define PHI_THREADS 192
#define PHI_ROWS    96     // 2 visits * 48 codes
#define RHO_THREADS 256
#define D128        128

// ---------------------------------------------------------------------------
// A-fragment loader: 16-bit A matrix 16x32 (MxK) layout per CDNA5 ISA 7.12.2.
// Lane l: row M = m-base + (l&15).  Element pairs (2j,2j+1) hold adjacent K
// values K = kt*32 + ((j>=4)?16:0) + (l>>4)*8 + (j&3)*2 (+0/+1) -> one u32.
// ---------------------------------------------------------------------------
__device__ __forceinline__ void load_a_frags(const __bf16* A, int arow,
                                             v16bf* afrag, int half) {
  const u32* base = (const u32*)(A + arow * D128);
#pragma unroll
  for (int kt = 0; kt < 4; ++kt) {
    u32* dst = (u32*)&afrag[kt];
#pragma unroll
    for (int j = 0; j < 8; ++j) {
      int k = kt * 32 + ((j >> 2) << 4) + half * 8 + ((j & 3) << 1);
      dst[j] = base[k >> 1];
    }
  }
}

// ---------------------------------------------------------------------------
// B-fragment loader: 16-bit B matrix 32x16 (KxN).  Lane l holds column N =
// n0 + (l&15); element s holds K = kbase + ((s>=8)?16:0) + (l>>4)*8 + (s&7)
// from a row-major [K][128] weight matrix in LDS (k arg = kbase + l).
// ---------------------------------------------------------------------------
__device__ __forceinline__ v16bf load_b_frag(const __bf16* W, int k, int n0) {
  v16bf b;
  const int kb = k & ~31, l = k & 31, h = l >> 4;
  const __bf16* src = W + n0 + (l & 15);
#pragma unroll
  for (int s = 0; s < 16; ++s)
    b[s] = src[(kb + ((s >> 3) << 4) + h * 8 + (s & 7)) * D128];
  return b;
}

__device__ __forceinline__ float silu_f(float x) {
  return x / (1.0f + __expf(-x));
}

// One 16-row M-tile through a full 128x128 GEMM (+bias, optional SiLU),
// result written back to LDS as bf16 row-major.
__device__ __forceinline__ void gemm_tile_pass(const __bf16* A, const __bf16* W,
                                               const float* bias, __bf16* Dst,
                                               int m0, int lane, bool do_silu) {
  const int half = lane >> 4, l16 = lane & 15;
  v16bf afrag[4];
  load_a_frags(A, m0 + l16, afrag, half);
#pragma unroll
  for (int nt = 0; nt < 8; ++nt) {
    const int n0 = nt << 4;
    const float bv = bias[n0 + l16];
    v8f acc = {bv, bv, bv, bv, bv, bv, bv, bv};
#pragma unroll
    for (int kt = 0; kt < 4; ++kt) {
      v16bf bfrag = load_b_frag(W, kt * 32 + lane, n0);
      acc = __builtin_amdgcn_wmma_f32_16x16x32_bf16(
          false, afrag[kt], false, bfrag, (short)0, acc, false, false);
    }
#pragma unroll
    for (int r = 0; r < 8; ++r) {
      float x = acc[r];
      if (do_silu) x = silu_f(x);
      Dst[(m0 + r + 8 * half) * D128 + n0 + l16] = (__bf16)x;
    }
  }
}

// Same but f32 output to global (final rho layer, no activation).
__device__ __forceinline__ void gemm_tile_pass_f32out(const __bf16* A,
                                                      const __bf16* W,
                                                      const float* bias,
                                                      float* Dst, int m0,
                                                      int lane) {
  const int half = lane >> 4, l16 = lane & 15;
  v16bf afrag[4];
  load_a_frags(A, m0 + l16, afrag, half);
#pragma unroll
  for (int nt = 0; nt < 8; ++nt) {
    const int n0 = nt << 4;
    const float bv = bias[n0 + l16];
    v8f acc = {bv, bv, bv, bv, bv, bv, bv, bv};
#pragma unroll
    for (int kt = 0; kt < 4; ++kt) {
      v16bf bfrag = load_b_frag(W, kt * 32 + lane, n0);
      acc = __builtin_amdgcn_wmma_f32_16x16x32_bf16(
          false, afrag[kt], false, bfrag, (short)0, acc, false, false);
    }
#pragma unroll
    for (int r = 0; r < 8; ++r)
      Dst[(m0 + r + 8 * half) * D128 + n0 + l16] = acc[r];
  }
}

// Attention logit pass: per-row sum over N of tanh(h2@Wa1+ba1)[.,n]*wa2[n].
// Accumulator tile columns live across the 16-lane halves -> xor-shuffle
// reduction (masks 1,2,4,8 stay within each half of the wave32).
__device__ __forceinline__ void attn_tile_pass(const __bf16* H, const __bf16* Wa,
                                               const float* bias,
                                               const float* wvec, float ba2v,
                                               float* logit, int m0, int lane) {
  const int half = lane >> 4, l16 = lane & 15;
  v16bf afrag[4];
  load_a_frags(H, m0 + l16, afrag, half);
  float rsum[8] = {0.f, 0.f, 0.f, 0.f, 0.f, 0.f, 0.f, 0.f};
#pragma unroll
  for (int nt = 0; nt < 8; ++nt) {
    const int n0 = nt << 4;
    const float bv = bias[n0 + l16];
    v8f acc = {bv, bv, bv, bv, bv, bv, bv, bv};
#pragma unroll
    for (int kt = 0; kt < 4; ++kt) {
      v16bf bfrag = load_b_frag(Wa, kt * 32 + lane, n0);
      acc = __builtin_amdgcn_wmma_f32_16x16x32_bf16(
          false, afrag[kt], false, bfrag, (short)0, acc, false, false);
    }
    const float w = wvec[n0 + l16];
#pragma unroll
    for (int r = 0; r < 8; ++r) rsum[r] += tanhf(acc[r]) * w;
  }
#pragma unroll
  for (int r = 0; r < 8; ++r) {
    float s = rsum[r];
    s += __shfl_xor(s, 1);
    s += __shfl_xor(s, 2);
    s += __shfl_xor(s, 4);
    s += __shfl_xor(s, 8);
    if (l16 == 0) logit[m0 + r + 8 * half] = s + ba2v;
  }
}

// ---------------------------------------------------------------------------
// Kernel 1: fused embedding-gather -> phi (2x Linear+SiLU) -> attention
// logits -> masked softmax -> attention pooling.  2 visits / iteration,
// 6 wave32s, persistent grid.  Weights cached in LDS as bf16.
// ---------------------------------------------------------------------------
__global__ __launch_bounds__(PHI_THREADS) void phi_attn_pool_kernel(
    const int* __restrict__ ids, const float* __restrict__ emb,
    const float* __restrict__ W1, const float* __restrict__ b1,
    const float* __restrict__ W2, const float* __restrict__ b2,
    const float* __restrict__ Wa1, const float* __restrict__ ba1,
    const float* __restrict__ wa2, const float* __restrict__ ba2,
    __bf16* __restrict__ hpool, int V, int nWG) {
  extern __shared__ char smem[];
  __bf16* W1s = (__bf16*)smem;
  __bf16* W2s = W1s + D128 * D128;
  __bf16* Was = W2s + D128 * D128;
  __bf16* bufA = Was + D128 * D128;        // x, then reused for h2
  __bf16* bufB = bufA + PHI_ROWS * D128;   // h1
  float* b1s = (float*)(bufB + PHI_ROWS * D128);
  float* b2s = b1s + D128;
  float* bas = b2s + D128;
  float* wa2s = bas + D128;
  float* logit = wa2s + D128;
  float* mrow = logit + PHI_ROWS;
  float* attw = mrow + PHI_ROWS;

  const int tid = threadIdx.x;
  const int wave = tid >> 5;
  const int lane = tid & 31;

  for (int i = tid; i < D128 * D128; i += PHI_THREADS) {
    W1s[i] = (__bf16)W1[i];
    W2s[i] = (__bf16)W2[i];
    Was[i] = (__bf16)Wa1[i];
  }
  if (tid < D128) {
    b1s[tid] = b1[tid];
    b2s[tid] = b2[tid];
    bas[tid] = ba1[tid];
    wa2s[tid] = wa2[tid];
  }
  const float ba2v = ba2[0];
  __syncthreads();

  const int pairs = V >> 1;
  for (int p = blockIdx.x; p < pairs; p += nWG) {
    const int v0 = p << 1;
    const int* idp = ids + v0 * 48;
    if (p + nWG < pairs)
      __builtin_prefetch(ids + (size_t)(p + nWG) * 2 * 48, 0, 1);

    // gather embedding rows (emb table is L2-resident: 10 MB << 192 MB)
    for (int i = tid; i < PHI_ROWS * D128; i += PHI_THREADS) {
      const int r = i >> 7, d = i & 127;
      const int id = idp[r];
      bufA[i] = (__bf16)emb[(size_t)id * D128 + d];
    }
    if (tid < PHI_ROWS) mrow[tid] = (idp[tid] != 0) ? 0.0f : -3.0e38f;
    __syncthreads();

    gemm_tile_pass(bufA, W1s, b1s, bufB, wave * 16, lane, true);  // h1
    __syncthreads();
    gemm_tile_pass(bufB, W2s, b2s, bufA, wave * 16, lane, true);  // h2 (over x)
    __syncthreads();
    attn_tile_pass(bufA, Was, bas, wa2s, ba2v, logit, wave * 16, lane);
    __syncthreads();

    if (tid < 2) {  // masked softmax over 48 codes per visit
      const int base = tid * 48;
      float mx = -3.4e38f;
      for (int c = 0; c < 48; ++c) mx = fmaxf(mx, logit[base + c] + mrow[base + c]);
      float s = 0.0f;
      for (int c = 0; c < 48; ++c) {
        const float e = __expf(logit[base + c] + mrow[base + c] - mx);
        attw[base + c] = e;
        s += e;
      }
      const float inv = 1.0f / s;
      for (int c = 0; c < 48; ++c) attw[base + c] *= inv;
    }
    __syncthreads();

    // attention-weighted pooling -> bf16 h_pool in workspace
    for (int i = tid; i < 2 * D128; i += PHI_THREADS) {
      const int v = i >> 7, d = i & 127;
      const __bf16* hv = bufA + v * 48 * D128;
      const float* wv = attw + v * 48;
      float acc = 0.0f;
      for (int c = 0; c < 48; ++c) acc += wv[c] * (float)hv[c * D128 + d];
      hpool[(size_t)(v0 + v) * D128 + d] = (__bf16)acc;
    }
    __syncthreads();
  }
}

// ---------------------------------------------------------------------------
// Kernel 2: rho head.  v = silu(h_pool@Wr1+br1)@Wr2+br2.  128 rows / block,
// 8 wave32s, one M-tile per wave, two WMMA passes through LDS.
// ---------------------------------------------------------------------------
__global__ __launch_bounds__(RHO_THREADS) void rho_kernel(
    const __bf16* __restrict__ hpool, const float* __restrict__ Wr1,
    const float* __restrict__ br1, const float* __restrict__ Wr2,
    const float* __restrict__ br2, float* __restrict__ out, int V) {
  extern __shared__ char smem[];
  __bf16* Wr1s = (__bf16*)smem;
  __bf16* Wr2s = Wr1s + D128 * D128;
  __bf16* hps = Wr2s + D128 * D128;
  __bf16* gs = hps + D128 * D128;
  float* br1s = (float*)(gs + D128 * D128);
  float* br2s = br1s + D128;

  const int tid = threadIdx.x;
  const int wave = tid >> 5;
  const int lane = tid & 31;

  for (int i = tid; i < D128 * D128; i += RHO_THREADS) {
    Wr1s[i] = (__bf16)Wr1[i];
    Wr2s[i] = (__bf16)Wr2[i];
  }
  if (tid < D128) {
    br1s[tid] = br1[tid];
    br2s[tid] = br2[tid];
  }
  const size_t row0 = (size_t)blockIdx.x * 128;
  for (int i = tid; i < 128 * D128; i += RHO_THREADS)
    hps[i] = hpool[row0 * D128 + i];
  __syncthreads();

  gemm_tile_pass(hps, Wr1s, br1s, gs, wave * 16, lane, true);
  __syncthreads();
  gemm_tile_pass_f32out(gs, Wr2s, br2s, out + row0 * D128, wave * 16, lane);
}

// ---------------------------------------------------------------------------
extern "C" void kernel_launch(void* const* d_in, const int* in_sizes, int n_in,
                              void* d_out, int out_size, void* d_ws,
                              size_t ws_size, hipStream_t stream) {
  const int* ids = (const int*)d_in[0];
  const float* emb = (const float*)d_in[1];
  const float* W1 = (const float*)d_in[2];
  const float* b1 = (const float*)d_in[3];
  const float* W2 = (const float*)d_in[4];
  const float* b2 = (const float*)d_in[5];
  const float* Wa1 = (const float*)d_in[6];
  const float* ba1 = (const float*)d_in[7];
  const float* wa2 = (const float*)d_in[8];
  const float* ba2 = (const float*)d_in[9];
  const float* Wr1 = (const float*)d_in[10];
  const float* br1 = (const float*)d_in[11];
  const float* Wr2 = (const float*)d_in[12];
  const float* br2 = (const float*)d_in[13];

  const int V = in_sizes[0] / 48;  // MAX_CODES = 48
  __bf16* hpool = (__bf16*)d_ws;   // V*128 bf16 = 4 MB

  const size_t phi_smem = (size_t)3 * D128 * D128 * 2 +
                          (size_t)2 * PHI_ROWS * D128 * 2 +
                          (4 * D128 + 3 * PHI_ROWS) * sizeof(float);
  const size_t rho_smem =
      (size_t)4 * D128 * D128 * 2 + 2 * D128 * sizeof(float);

  (void)hipFuncSetAttribute(reinterpret_cast<const void*>(phi_attn_pool_kernel),
                            hipFuncAttributeMaxDynamicSharedMemorySize,
                            (int)phi_smem);
  (void)hipFuncSetAttribute(reinterpret_cast<const void*>(rho_kernel),
                            hipFuncAttributeMaxDynamicSharedMemorySize,
                            (int)rho_smem);

  const int pairs = V >> 1;
  const int nWG = pairs < 512 ? pairs : 512;

  phi_attn_pool_kernel<<<nWG, PHI_THREADS, phi_smem, stream>>>(
      ids, emb, W1, b1, W2, b2, Wa1, ba1, wa2, ba2, hpool, V, nWG);

  const int rho_blocks = V / 128;
  rho_kernel<<<rho_blocks, RHO_THREADS, rho_smem, stream>>>(
      hpool, Wr1, br1, Wr2, br2, (float*)d_out, V);
}
